// GAFN_V2_19292993093865
// MI455X (gfx1250) — matrix operand order repaired and checked by reading, not compiled, not run on hardware
//
#include <hip/hip_runtime.h>
#include <hip/hip_bf16.h>
#include <math.h>

// Problem constants from reference: N=8, C=1024, D=128, K=64, R=256
#define NN 8
#define CC 1024
#define DD 128
#define KK 64
#define RR 256
#define EPSV 1e-12f

typedef __attribute__((ext_vector_type(16))) _Float16 v16h;
typedef __attribute__((ext_vector_type(8)))  float    v8f;
typedef __attribute__((ext_vector_type(4)))  float    v4f;   // native vec for NT stores

// ---------------------------------------------------------------------------
// Kernel 1: L2-normalize each x row (N*C rows of D=128 floats).
// One wave (32 lanes) per row, float4 per lane, wave reduction via shfl_xor.
// ---------------------------------------------------------------------------
__global__ __launch_bounds__(256) void nv_norm_x(const float* __restrict__ x,
                                                 float* __restrict__ xn) {
  int row  = blockIdx.x * 8 + (threadIdx.x >> 5);   // [0, N*C)
  int lane = threadIdx.x & 31;
  const float4* src = (const float4*)(x + (size_t)row * DD);
  float4 v = src[lane];
  float ss = v.x * v.x + v.y * v.y + v.z * v.z + v.w * v.w;
#pragma unroll
  for (int m = 16; m >= 1; m >>= 1) ss += __shfl_xor(ss, m, 32);
  float inv = 1.0f / fmaxf(sqrtf(ss), EPSV);
  float4 o = {v.x * inv, v.y * inv, v.z * inv, v.w * inv};
  ((float4*)(xn + (size_t)row * DD))[lane] = o;
}

// ---------------------------------------------------------------------------
// Kernel 2: gather cent[n][k][:] = cluster[n][cluster_idx[k]][:]
//           and centnorm2[n][k] = ||cent||^2.  One wave per (n,k) row.
// ---------------------------------------------------------------------------
__global__ __launch_bounds__(256) void nv_gather_cent(const float* __restrict__ cluster,
                                                      const int* __restrict__ cidx,
                                                      float* __restrict__ cent,
                                                      float* __restrict__ cn2) {
  int row  = blockIdx.x * 8 + (threadIdx.x >> 5);   // [0, N*K)
  int lane = threadIdx.x & 31;
  int n = row >> 6;
  int k = row & (KK - 1);
  int r = cidx[k];
  const float4* src = (const float4*)(cluster + ((size_t)n * RR + r) * DD);
  float4 v = src[lane];
  ((float4*)(cent + (size_t)row * DD))[lane] = v;
  float ss = v.x * v.x + v.y * v.y + v.z * v.z + v.w * v.w;
#pragma unroll
  for (int m = 16; m >= 1; m >>= 1) ss += __shfl_xor(ss, m, 32);
  if (lane == 0) cn2[row] = ss;
}

// ---------------------------------------------------------------------------
// Kernel 3: WMMA dual-GEMM + softmax + fused normalization scale.
// One wave per (n, 16-row c-tile).  For the 16 c-rows it computes
//   logits[16][64] = x_hat @ w^T  (+bias)   and   dot[16][64] = x_hat @ cent^T
// with v_wmma_f32_16x16x32_f16 (4 k-tiles x 4 d-chunks x 2 = 32 WMMAs),
// then per c-row: softmax over K, first-norm factor from
// ||x-cent||^2 = 1 + ||cent||^2 - 2*dot, second norm over the 64 blocks,
// emitting scale[n][c][k] so the store kernel is a single multiply.
// ---------------------------------------------------------------------------
__global__ __launch_bounds__(32) void nv_scale(const float* __restrict__ xn,
                                               const float* __restrict__ w,
                                               const float* __restrict__ bias,
                                               const float* __restrict__ cent,
                                               const float* __restrict__ cn2,
                                               float* __restrict__ scale) {
  __shared__ float l_logit[16][KK];
  __shared__ float l_dot[16][KK];

  int l = threadIdx.x;
  int m = l & 15;     // A-row / B-col / D-col index per ISA layout
  int h = l >> 4;     // half-wave select
  int bid = blockIdx.x;           // [0, N * C/16)
  int n  = bid >> 6;
  int c0 = (bid & 63) * 16;

  // ---- A fragments: x_hat rows, 16x32 f16 per d-chunk (K = 16*h + j) --
  const float* xrow = xn + ((size_t)(n * CC + c0 + m)) * DD;
  v16h afr[4];
#pragma unroll
  for (int cc = 0; cc < 4; ++cc) {
    int db = cc * 32;
#pragma unroll
    for (int j = 0; j < 8; ++j) afr[cc][j]     = (_Float16)xrow[db + 16 * h + j];
#pragma unroll
    for (int j = 0; j < 8; ++j) afr[cc][8 + j] = (_Float16)xrow[db + 16 * h + 8 + j];
  }

#pragma unroll
  for (int kt = 0; kt < 4; ++kt) {
    int k = kt * 16 + m;                       // this lane's output column
    const float* wrow = w    + (size_t)k * DD;
    const float* crow = cent + ((size_t)(n * KK + k)) * DD;
    v8f cl = {};
    v8f cd = {};
#pragma unroll
    for (int cc = 0; cc < 4; ++cc) {
      int db = cc * 32;
      v16h bw, bc;
#pragma unroll
      for (int j = 0; j < 16; ++j) {
        bw[j] = (_Float16)wrow[db + 16 * h + j];   // B: K = 16*h + j, N = m
        bc[j] = (_Float16)crow[db + 16 * h + j];
      }
      cl = __builtin_amdgcn_wmma_f32_16x16x32_f16(false, afr[cc], false, bw,
                                                  (short)0, cl, false, false);
      cd = __builtin_amdgcn_wmma_f32_16x16x32_f16(false, afr[cc], false, bc,
                                                  (short)0, cd, false, false);
    }
    float bk = bias[k];
#pragma unroll
    for (int i = 0; i < 8; ++i) {              // D row = i + 8*h, col = k
      l_logit[i + 8 * h][k] = cl[i] + bk;
      l_dot[i + 8 * h][k]   = cd[i];
    }
  }
  __syncthreads();

  // ---- per-row softmax + fused two-stage normalization --------------------
  if (l < 16) {
    int row = l;
    int c   = c0 + row;
    float mx = -3.4e38f;
#pragma unroll 8
    for (int k = 0; k < KK; ++k) mx = fmaxf(mx, l_logit[row][k]);
    float se = 0.0f;
#pragma unroll 8
    for (int k = 0; k < KK; ++k) se += __expf(l_logit[row][k] - mx);
    float ise = 1.0f / se;
    float g2 = 0.0f;
#pragma unroll 8
    for (int k = 0; k < KK; ++k) {
      float s  = __expf(l_logit[row][k] - mx) * ise;           // softmax
      float d2 = fmaxf(1.0f + cn2[n * KK + k] - 2.0f * l_dot[row][k], 0.0f);
      float rn = s * sqrtf(d2);                                // ||residual||
      float f1 = 1.0f / fmaxf(rn, EPSV);                       // 1st l2norm
      float g  = rn * f1;                                      // block norm
      g2 += g * g;
      l_logit[row][k] = s * f1;                                // stash s*f1
    }
    float f2 = 1.0f / fmaxf(sqrtf(g2), EPSV);                  // 2nd l2norm
    float* so = scale + ((size_t)(n * CC + c)) * KK;
#pragma unroll 8
    for (int k = 0; k < KK; ++k) so[k] = l_logit[row][k] * f2;
  }
}

// ---------------------------------------------------------------------------
// Kernel 4: the 256 MB streaming-store kernel (the roofline term).
// Each block: one n, 4 consecutive c rows. Centroid table (64x128 = 32 KB)
// lives in LDS (320 KB budget), reused across the 4 rows.
// out[n][c][k*128+d] = (x_hat[d] - cent[k][d]) * scale[k].
// Output (256 MB) exceeds the 192 MB L2 and is never re-read, so emit the
// stores with the non-temporal hint (TH=NT) to avoid churning L2.
// ---------------------------------------------------------------------------
__global__ __launch_bounds__(256) void nv_store(const float* __restrict__ xn,
                                                const float* __restrict__ cent,
                                                const float* __restrict__ scale,
                                                float* __restrict__ out) {
  __shared__ float cent_s[KK * DD];   // 32 KB
  __shared__ float x_s[DD];
  __shared__ float sc_s[KK];

  int t   = threadIdx.x;
  int bid = blockIdx.x;               // [0, N * C/4)
  int n   = bid >> 8;
  int cb  = (bid & 255) * 4;

  const float4* csrc = (const float4*)(cent + (size_t)n * KK * DD);
  float4* cdst = (float4*)cent_s;
#pragma unroll
  for (int i = 0; i < 8; ++i) cdst[t + i * 256] = csrc[t + i * 256];

  for (int c = cb; c < cb + 4; ++c) {
    __syncthreads();
    if (t < 32)
      ((float4*)x_s)[t] = ((const float4*)(xn + ((size_t)(n * CC + c)) * DD))[t];
    if (t < KK)
      sc_s[t] = scale[((size_t)(n * CC + c)) * KK + t];
    __syncthreads();

    float* ob = out + ((size_t)(n * CC + c)) * (KK * DD);
#pragma unroll
    for (int i = 0; i < 8; ++i) {
      int e = i * 1024 + t * 4;       // flat index into the 8192-wide row
      int k = e >> 7;
      int d = e & (DD - 1);
      float s  = sc_s[k];
      v4f  xv = *(const v4f*)(x_s + d);
      v4f  cv = *(const v4f*)(cent_s + k * DD + d);
      v4f  r  = (xv - cv) * s;
      __builtin_nontemporal_store(r, (v4f*)(ob + e));
    }
  }
}

// ---------------------------------------------------------------------------
extern "C" void kernel_launch(void* const* d_in, const int* in_sizes, int n_in,
                              void* d_out, int out_size, void* d_ws, size_t ws_size,
                              hipStream_t stream) {
  const float* x       = (const float*)d_in[0];   // (N,C,D)
  const float* cluster = (const float*)d_in[1];   // (N,R,D)
  const float* conv_w  = (const float*)d_in[2];   // (K,D)
  const float* conv_b  = (const float*)d_in[3];   // (K,)
  const int*   cidx    = (const int*)d_in[4];     // (K,)
  float* out = (float*)d_out;                     // (N,C,K*D)

  // workspace carve-out (~6.6 MB)
  float* xn    = (float*)d_ws;                    // N*C*D   = 1048576
  float* cent  = xn   + (size_t)NN * CC * DD;     // N*K*D   = 65536
  float* cn2   = cent + (size_t)NN * KK * DD;     // N*K     = 512
  float* scale = cn2  + (size_t)NN * KK;          // N*C*K   = 524288

  nv_norm_x<<<(NN * CC) / 8, 256, 0, stream>>>(x, xn);
  nv_gather_cent<<<(NN * KK) / 8, 256, 0, stream>>>(cluster, cidx, cent, cn2);
  nv_scale<<<NN * (CC / 16), 32, 0, stream>>>(xn, conv_w, conv_b, cent, cn2, scale);
  nv_store<<<NN * (CC / 4), 256, 0, stream>>>(xn, cent, scale, out);
}
